// BahdanauAttention_63144609185923
// MI455X (gfx1250) — matrix operand order repaired and checked by reading, not compiled, not run on hardware
//
#include <hip/hip_runtime.h>
#include <hip/hip_bf16.h>
#include <math.h>

typedef __attribute__((ext_vector_type(16))) __bf16 v16bf;
typedef __attribute__((ext_vector_type(8)))  float  v8f;
typedef __attribute__((ext_vector_type(4)))  float  v4f;
typedef __bf16 bf16;

#define ENC 2048
#define DEC 512
#define ATT 512
#define NB  64
#define NL  1024

// ---------------------------------------------------------------------------
// K0a: W1 [ENC, ATT] f32  ->  W1T [ATT, ENC] bf16  (LDS-tiled transpose)
// ---------------------------------------------------------------------------
__global__ void __launch_bounds__(256) k_w1_transpose(const float* __restrict__ W1,
                                                      bf16* __restrict__ W1T) {
    __shared__ float tile[32][33];
    const int tx = threadIdx.x & 31;
    const int ty = threadIdx.x >> 5;          // 0..7
    const int er = blockIdx.x * 32;           // ENC base (blockIdx.x: 0..63)
    const int ac = blockIdx.y * 32;           // ATT base (blockIdx.y: 0..15)
#pragma unroll
    for (int i = 0; i < 4; ++i) {
        int r = ty + i * 8;
        tile[r][tx] = W1[(size_t)(er + r) * ATT + (ac + tx)];
    }
    __syncthreads();
#pragma unroll
    for (int i = 0; i < 4; ++i) {
        int a = ac + ty + i * 8;
        W1T[(size_t)a * ENC + (er + tx)] = (bf16)tile[tx][ty + i * 8];
    }
}

// ---------------------------------------------------------------------------
// K0b: proj_h[b][a] = hidden[b] . W2[:,a] + b2[a]     (tiny, 33 MFLOP)
// ---------------------------------------------------------------------------
__global__ void __launch_bounds__(256) k_proj_h(const float* __restrict__ hs,
                                                const float* __restrict__ W2,
                                                const float* __restrict__ b2,
                                                float* __restrict__ projh) {
    const int idx = blockIdx.x * 256 + threadIdx.x;   // 0..32767
    const int b = idx >> 9;
    const int a = idx & (ATT - 1);
    const float* h = hs + (size_t)b * DEC;
    float acc = b2[a];
    for (int d = 0; d < DEC; ++d)
        acc = fmaf(h[d], W2[(size_t)d * ATT + a], acc);
    projh[idx] = acc;
}

// ---------------------------------------------------------------------------
// K1: fused  tanh(features@W1 + b1 + proj_h) @ V + bv  -> raw scores [B, L]
//     grid (L/16, B), 256 thr = 8 waves, wave w owns ATT columns [64w, 64w+64)
//     bf16 WMMA 16x16x32, f32 accumulate.
// ---------------------------------------------------------------------------
__global__ void __launch_bounds__(256) k_score_wmma(
    const float* __restrict__ features, const bf16* __restrict__ W1T,
    const float* __restrict__ projh, const float* __restrict__ b1,
    const float* __restrict__ V, const float* __restrict__ bv,
    float* __restrict__ scores) {

    const int b    = blockIdx.y;
    const int lt   = blockIdx.x;          // l-tile, 16 rows
    const int tid  = threadIdx.x;
    const int lane = tid & 31;
    const int wave = tid >> 5;            // 0..7
    const int half = lane >> 4;           // 0|1
    const int l15  = lane & 15;

    __shared__ float partial[8 * 16];

    // A fragment addressing (contiguous k order, same as B):
    //   lane<16 : M=lane,    K = kk+0..15
    //   lane>=16: M=lane-16, K = kk+16..31
    const float* ap = features + ((size_t)b * NL + (size_t)lt * 16 + l15) * ENC
                      + half * 16;
    // B fragment addressing (32x16 bf16): lane holds column n = tile_base+l15,
    //   lanes 0-15: K = kk+0..15, lanes 16-31: K = kk+16..31  (contiguous in W1T)
    const bf16* brow[4];
#pragma unroll
    for (int t = 0; t < 4; ++t)
        brow[t] = W1T + (size_t)(wave * 64 + t * 16 + l15) * ENC + half * 16;

    v8f acc[4] = {};

    for (int kk = 0; kk < ENC; kk += 32) {
        v4f x0 = *(const v4f*)(ap + kk);
        v4f x1 = *(const v4f*)(ap + kk + 4);
        v4f x2 = *(const v4f*)(ap + kk + 8);
        v4f x3 = *(const v4f*)(ap + kk + 12);
        v16bf a;
        a[0]  = (bf16)x0[0]; a[1]  = (bf16)x0[1]; a[2]  = (bf16)x0[2]; a[3]  = (bf16)x0[3];
        a[4]  = (bf16)x1[0]; a[5]  = (bf16)x1[1]; a[6]  = (bf16)x1[2]; a[7]  = (bf16)x1[3];
        a[8]  = (bf16)x2[0]; a[9]  = (bf16)x2[1]; a[10] = (bf16)x2[2]; a[11] = (bf16)x2[3];
        a[12] = (bf16)x3[0]; a[13] = (bf16)x3[1]; a[14] = (bf16)x3[2]; a[15] = (bf16)x3[3];

#pragma unroll
        for (int t = 0; t < 4; ++t) {
            v16bf bf = *(const v16bf*)(brow[t] + kk);
            acc[t] = __builtin_amdgcn_wmma_f32_16x16x32_bf16(
                false, a, false, bf, (short)0, acc[t], false, false);
        }
    }

    // Epilogue: bias + tanh + dot with V.
    // acc[t][v] is element (M = v + 8*half, N = wave*64 + t*16 + l15).
    float csum[8] = {0.f, 0.f, 0.f, 0.f, 0.f, 0.f, 0.f, 0.f};
#pragma unroll
    for (int t = 0; t < 4; ++t) {
        const int n = wave * 64 + t * 16 + l15;
        const float bias = b1[n] + projh[(size_t)b * ATT + n];
        const float vs = V[n];
#pragma unroll
        for (int v = 0; v < 8; ++v)
            csum[v] += tanhf(acc[t][v] + bias) * vs;
    }
    // Deterministic reduce over the 16 lanes of each half (same M per half).
#pragma unroll
    for (int v = 0; v < 8; ++v) {
        float s = csum[v];
        s += __shfl_xor(s, 1, 32);
        s += __shfl_xor(s, 2, 32);
        s += __shfl_xor(s, 4, 32);
        s += __shfl_xor(s, 8, 32);
        csum[v] = s;
    }
    if (l15 == 0) {
#pragma unroll
        for (int v = 0; v < 8; ++v)
            partial[wave * 16 + half * 8 + v] = csum[v];
    }
    __syncthreads();
    if (tid < 16) {
        float s = bv[0];
#pragma unroll
        for (int w = 0; w < 8; ++w) s += partial[w * 16 + tid];
        scores[(size_t)b * NL + (size_t)lt * 16 + tid] = s;
    }
}

// ---------------------------------------------------------------------------
// K2: in-place softmax over L=1024 per batch (one workgroup per batch)
// ---------------------------------------------------------------------------
__global__ void __launch_bounds__(256) k_softmax(float* __restrict__ w) {
    const int b = blockIdx.x;
    const int tid = threadIdx.x;
    __shared__ float red[256];
    float* wb = w + (size_t)b * NL;
    float v[4];
#pragma unroll
    for (int i = 0; i < 4; ++i) v[i] = wb[tid + i * 256];
    float m = fmaxf(fmaxf(v[0], v[1]), fmaxf(v[2], v[3]));
    red[tid] = m;
    __syncthreads();
    for (int s = 128; s > 0; s >>= 1) {
        if (tid < s) red[tid] = fmaxf(red[tid], red[tid + s]);
        __syncthreads();
    }
    m = red[0];
    __syncthreads();
    float e[4], sum = 0.f;
#pragma unroll
    for (int i = 0; i < 4; ++i) { e[i] = __expf(v[i] - m); sum += e[i]; }
    red[tid] = sum;
    __syncthreads();
    for (int s = 128; s > 0; s >>= 1) {
        if (tid < s) red[tid] += red[tid + s];
        __syncthreads();
    }
    const float inv = 1.0f / red[0];
#pragma unroll
    for (int i = 0; i < 4; ++i) wb[tid + i * 256] = e[i] * inv;
}

// ---------------------------------------------------------------------------
// K3: context[b][e] = sum_l w[b][l] * features[b][l][e]   (bandwidth-bound)
//     grid (2, B): chunk of 1024 ENC floats per workgroup, float4 per thread
// ---------------------------------------------------------------------------
__global__ void __launch_bounds__(256) k_context(const float* __restrict__ features,
                                                 const float* __restrict__ w,
                                                 float* __restrict__ ctx) {
    const int b = blockIdx.y;
    const int e0 = blockIdx.x * 1024 + threadIdx.x * 4;
    __shared__ float lw[256];
    v4f acc = {};
    const float* fb = features + (size_t)b * NL * ENC + e0;
    const float* wb = w + (size_t)b * NL;
    for (int l0 = 0; l0 < NL; l0 += 256) {
        __syncthreads();
        lw[threadIdx.x] = wb[l0 + threadIdx.x];
        __syncthreads();
#pragma unroll 4
        for (int j = 0; j < 256; ++j) {
            v4f f = *(const v4f*)(fb + (size_t)(l0 + j) * ENC);
            acc += lw[j] * f;
        }
    }
    *(v4f*)(ctx + (size_t)b * ENC + e0) = acc;
}

// ---------------------------------------------------------------------------
extern "C" void kernel_launch(void* const* d_in, const int* in_sizes, int n_in,
                              void* d_out, int out_size, void* d_ws, size_t ws_size,
                              hipStream_t stream) {
    (void)in_sizes; (void)n_in; (void)out_size; (void)ws_size;

    const float* features = (const float*)d_in[0];   // [64,1024,2048]
    const float* hidden   = (const float*)d_in[1];   // [64,512]
    const float* W1       = (const float*)d_in[2];   // [2048,512]
    const float* b1       = (const float*)d_in[3];   // [512]
    const float* W2       = (const float*)d_in[4];   // [512,512]
    const float* b2       = (const float*)d_in[5];   // [512]
    const float* V        = (const float*)d_in[6];   // [512,1]
    const float* bv       = (const float*)d_in[7];   // [1]

    float* out = (float*)d_out;
    float* ctx = out;                         // [64,2048] context_vector
    float* wts = out + (size_t)NB * ENC;      // [64,1024] scores -> weights

    char*  ws    = (char*)d_ws;
    bf16*  W1T   = (bf16*)ws;                                         // 2 MB
    float* projh = (float*)(ws + (size_t)ATT * ENC * sizeof(bf16));   // 128 KB

    k_w1_transpose<<<dim3(ENC / 32, ATT / 32), 256, 0, stream>>>(W1, W1T);
    k_proj_h<<<dim3((NB * ATT) / 256), 256, 0, stream>>>(hidden, W2, b2, projh);
    k_score_wmma<<<dim3(NL / 16, NB), 256, 0, stream>>>(features, W1T, projh,
                                                        b1, V, bv, wts);
    k_softmax<<<dim3(NB), 256, 0, stream>>>(wts);
    k_context<<<dim3(2, NB), 256, 0, stream>>>(features, wts, ctx);
}
